// FlowExtract2D_83769042141669
// MI455X (gfx1250) — matrix operand order repaired and checked by reading, not compiled, not run on hardware
//
#include <hip/hip_runtime.h>
#include <hip/hip_bf16.h>
#include <math.h>

typedef __attribute__((ext_vector_type(16))) _Float16 v16h;
typedef __attribute__((ext_vector_type(8)))  _Float16 v8h;
typedef __attribute__((ext_vector_type(8)))  float    v8f;

#define MD    20
#define NB    4
#define NC    64
#define NH    112
#define NW    112
#define HP    (NH + 2*MD)   /* 152 padded rows */
#define WPAD  160           /* padded width >= 152, tail zeros for tile overrun */
#define KD    (2*MD + 1)    /* 41 */
#define TINV  1.0e4f        /* 1 / softmax temperature */
#define NTILE 7             /* 112 / 16 */
#define LS    66            /* LDS row stride (floats), conflict-free */

// ---------------- normalize in1 -> f16 [B][H][W][C] ----------------
__global__ __launch_bounds__(256)
void flowx_norm1_kernel(const float* __restrict__ in, _Float16* __restrict__ out) {
  int idx = blockIdx.x * blockDim.x + threadIdx.x;
  if (idx >= NB*NH*NW) return;
  int w = idx % NW, h = (idx / NW) % NH, b = idx / (NW*NH);
  const float* src = in + ((size_t)b*NC*NH + h)*NW + w;   // + c*NH*NW per channel
  float v[NC]; float s = 0.f;
  #pragma unroll
  for (int c = 0; c < NC; ++c) { float x = src[(size_t)c*NH*NW]; v[c] = x; s += x*x; }
  float inv = 1.f / fmaxf(sqrtf(s), 1e-12f);
  _Float16* dst = out + (size_t)idx * NC;
  #pragma unroll
  for (int c = 0; c < NC; ++c) dst[c] = (_Float16)(v[c] * inv);
}

// -------- normalize in2 -> f16 zero-padded [B][HP][WPAD][C] --------
__global__ __launch_bounds__(256)
void flowx_norm2_kernel(const float* __restrict__ in, _Float16* __restrict__ out) {
  int idx = blockIdx.x * blockDim.x + threadIdx.x;
  if (idx >= NB*HP*WPAD) return;
  int wp = idx % WPAD, hp = (idx / WPAD) % HP, b = idx / (WPAD*HP);
  _Float16* dst = out + (size_t)idx * NC;
  int h = hp - MD, w = wp - MD;
  if ((unsigned)h < (unsigned)NH && (unsigned)w < (unsigned)NW) {
    const float* src = in + ((size_t)b*NC*NH + h)*NW + w;
    float v[NC]; float s = 0.f;
    #pragma unroll
    for (int c = 0; c < NC; ++c) { float x = src[(size_t)c*NH*NW]; v[c] = x; s += x*x; }
    float inv = 1.f / fmaxf(sqrtf(s), 1e-12f);
    #pragma unroll
    for (int c = 0; c < NC; ++c) dst[c] = (_Float16)(v[c] * inv);
  } else {
    v8h z = {};
    #pragma unroll
    for (int c = 0; c < NC; c += 8) *(v8h*)(dst + c) = z;
  }
}

// ---- helpers for the fused correlation kernel ----

// Load the 8 B fragments (4 N-tiles x 2 K-chunks) for one shifted row.
// B fragment layout (32x16 f16): lanes 0-15 hold N=lm, K={0..7,16..23}; lanes 16-31 K={8..15,24..31}.
__device__ __forceinline__
void flowx_load_b(const _Float16* __restrict__ brow, int w0, int lm, int kbase,
                  v16h* __restrict__ Bf) {
  #pragma unroll
  for (int t = 0; t < 4; ++t) {
    const _Float16* bp = brow + (size_t)(w0 + t*16 + lm) * NC;
    #pragma unroll
    for (int kc = 0; kc < 2; ++kc) {
      const _Float16* q = bp + kc*32 + (kbase >> 1);
      v8h lo = *(const v8h*)(q);
      v8h hi = *(const v8h*)(q + 16);
      Bf[t*2 + kc] =
          __builtin_shufflevector(lo, hi, 0,1,2,3,4,5,6,7,8,9,10,11,12,13,14,15);
    }
  }
}

// 8 WMMAs for the 16x64 band tile, then stage D into LDS.
// D layout: lane holds col N=lm(+16t), rows M = r + 8*half.
__device__ __forceinline__
void flowx_wmma_store(const v16h* __restrict__ A, const v16h* __restrict__ Bf,
                      float* __restrict__ lds, int half, int lm) {
  v8f acc[4] = {};
  #pragma unroll
  for (int t = 0; t < 4; ++t) {
    #pragma unroll
    for (int kc = 0; kc < 2; ++kc) {
      acc[t] = __builtin_amdgcn_wmma_f32_16x16x32_f16(
          /*neg_a=*/false, A[kc], /*neg_b=*/false, Bf[t*2 + kc],
          /*c_mod=*/(short)0, acc[t], /*reuse_a=*/false, /*reuse_b=*/false);
    }
  }
  #pragma unroll
  for (int t = 0; t < 4; ++t) {
    #pragma unroll
    for (int r = 0; r < 8; ++r)
      lds[(r + half*8)*LS + t*16 + lm] = acc[t][r];
  }
}

// Online-softmax accumulation over this dx's dy values for pixel p=lm.
// (single wave32 workgroup: LDS ops from one wave complete in order -> no barrier)
__device__ __forceinline__
void flowx_softmax_accum(const float* __restrict__ lds, int lm, int d0, int dn,
                         float dxv, float& mx, float& sm, float& fx, float& fy) {
  for (int i = 0; i < dn; ++i) {
    int d = d0 + i;
    float s = lds[lm*LS + lm + d];
    float dyv = (float)(d - MD);
    if (s > mx) {
      float sc = __expf((mx - s) * TINV);
      sm = sm*sc + 1.f;
      fx = fx*sc + dxv;
      fy = fy*sc + dyv;
      mx = s;
    } else {
      float e = __expf((s - mx) * TINV);
      sm += e; fx += e*dxv; fy += e*dyv;
    }
  }
}

// ---- fused correlation (WMMA band-GEMM) + online softmax + flow ----
// one wave32 per (b, h, 16-pixel tile); ping-pong pipelined over 41 row shifts.
__global__ __launch_bounds__(32)
void flowx_corr_flow_kernel(const _Float16* __restrict__ a1,
                            const _Float16* __restrict__ a2,
                            float* __restrict__ flow) {
  __shared__ float lds[16 * LS];

  int tile = blockIdx.x % NTILE;
  int h    = (blockIdx.x / NTILE) % NH;
  int b    = blockIdx.x / (NTILE * NH);
  int w0   = tile * 16;
  int lane = threadIdx.x;
  int half = lane >> 4;             // 0: lanes 0-15, 1: lanes 16-31
  int lm   = lane & 15;
  int kbase = half * 16;

  // A fragments (16x32 f16, ISA layout): lane<16 holds M=lm, K={0..7,16..23}+32*kc;
  // lanes>=16 hold M=lm, K={8..15,24..31}+32*kc.
  v16h A[2];
  {
    const _Float16* ap = a1 + ((size_t)((b*NH + h)*NW) + (w0 + lm)) * NC;
    #pragma unroll
    for (int kc = 0; kc < 2; ++kc) {
      int kb = kc*32 + half*8;
      v8h lo = *(const v8h*)(ap + kb);
      v8h hi = *(const v8h*)(ap + kb + 16);
      A[kc] = __builtin_shufflevector(lo, hi, 0,1,2,3,4,5,6,7,8,9,10,11,12,13,14,15);
    }
  }

  const _Float16* brow0 = a2 + ((size_t)(b*HP + h)) * WPAD * NC;  // dxi stride below
  const size_t rstride = (size_t)WPAD * NC;

  // online softmax state: pixel p = lm, dy-indices [d0, d0+dn)
  float mx = -3.4e38f, sm = 0.f, fx = 0.f, fy = 0.f;
  const int d0 = half * 21;
  const int dn = 21 - half;

  v16h B0[8], B1[8];
  flowx_load_b(brow0, w0, lm, kbase, B0);          // prologue: dxi = 0

  int dxi = 0;
  while (dxi < KD) {
    // ---- even step: consume B0, preload dxi+1 into B1 ----
    flowx_wmma_store(A, B0, lds, half, lm);
    if (dxi + 1 < KD)
      flowx_load_b(brow0 + (size_t)(dxi + 1)*rstride, w0, lm, kbase, B1);
    if (dxi + 2 < KD)   // gfx1250 global_prefetch two row-shifts ahead
      __builtin_prefetch(brow0 + (size_t)(dxi + 2)*rstride + (size_t)(w0 + lane)*NC, 0, 1);
    flowx_softmax_accum(lds, lm, d0, dn, (float)(dxi - MD), mx, sm, fx, fy);
    ++dxi;
    if (dxi >= KD) break;

    // ---- odd step: consume B1, preload dxi+1 into B0 ----
    flowx_wmma_store(A, B1, lds, half, lm);
    if (dxi + 1 < KD)
      flowx_load_b(brow0 + (size_t)(dxi + 1)*rstride, w0, lm, kbase, B0);
    if (dxi + 2 < KD)
      __builtin_prefetch(brow0 + (size_t)(dxi + 2)*rstride + (size_t)(w0 + lane)*NC, 0, 1);
    flowx_softmax_accum(lds, lm, d0, dn, (float)(dxi - MD), mx, sm, fx, fy);
    ++dxi;
  }

  // merge the two dy-halves (lane p <-> lane p+16)
  float mx2 = __shfl_xor(mx, 16, 32);
  float sm2 = __shfl_xor(sm, 16, 32);
  float fx2 = __shfl_xor(fx, 16, 32);
  float fy2 = __shfl_xor(fy, 16, 32);
  float M  = fmaxf(mx, mx2);
  float e1 = __expf((mx  - M) * TINV);
  float e2 = __expf((mx2 - M) * TINV);
  float Sm = sm*e1 + sm2*e2;
  float Fx = fx*e1 + fx2*e2;
  float Fy = fy*e1 + fy2*e2;
  if (half == 0) {
    int w = w0 + lm;
    flow[((size_t)(b*2 + 0)*NH + h)*NW + w] = Fx / Sm;  // guide[:,0] = dx (row shift)
    flow[((size_t)(b*2 + 1)*NH + h)*NW + w] = Fy / Sm;  // guide[:,1] = dy (col shift)
  }
}

extern "C" void kernel_launch(void* const* d_in, const int* in_sizes, int n_in,
                              void* d_out, int out_size, void* d_ws, size_t ws_size,
                              hipStream_t stream) {
  const float* in1 = (const float*)d_in[0];
  const float* in2 = (const float*)d_in[1];
  float* flow = (float*)d_out;

  char* ws = (char*)d_ws;
  size_t n1_bytes = (size_t)NB*NH*NW*NC * sizeof(_Float16);
  n1_bytes = (n1_bytes + 255) & ~(size_t)255;
  _Float16* n1 = (_Float16*)ws;                 // [B][H][W][C]        ~6.4 MB
  _Float16* n2 = (_Float16*)(ws + n1_bytes);    // [B][HP][WPAD][C]   ~12.5 MB

  int t1 = NB*NH*NW;
  hipLaunchKernelGGL(flowx_norm1_kernel, dim3((t1 + 255)/256), dim3(256), 0, stream, in1, n1);
  int t2 = NB*HP*WPAD;
  hipLaunchKernelGGL(flowx_norm2_kernel, dim3((t2 + 255)/256), dim3(256), 0, stream, in2, n2);
  hipLaunchKernelGGL(flowx_corr_flow_kernel, dim3(NB*NH*NTILE), dim3(32), 0, stream, n1, n2, flow);
}
